// DVGAE_26414048870608
// MI455X (gfx1250) — matrix operand order repaired and checked by reading, not compiled, not run on hardware
//
#include <hip/hip_runtime.h>
#include <hip/hip_bf16.h>

#define N_NODESC 10000
#define N_EDGESC 320000
#define IN_DIMC  512
#define CHC      128

typedef __bf16 bf16;
typedef __attribute__((ext_vector_type(16))) __bf16 v16bf;
typedef __attribute__((ext_vector_type(8)))  __bf16 v8bf;
typedef __attribute__((ext_vector_type(8)))  float  v8f;

// ---------------- f32 -> bf16 conversion ----------------
__global__ void cvt_bf16_kernel(const float* __restrict__ in, bf16* __restrict__ out, int n) {
    int i = blockIdx.x * 256 + threadIdx.x;
    if (i < n) out[i] = (bf16)in[i];
}

// ---------------- WMMA GEMM: D[M,N] = A[M,K] * B[N,K]^T (both row-major bf16) ----------------
// One 16x16 output tile per wave; 4 waves per block. M%16==0, N%16==0, K%32==0.
__global__ void wmma_gemm_kernel(const bf16* __restrict__ A, const bf16* __restrict__ B,
                                 float* __restrict__ D, int M, int N, int K) {
    int lane   = threadIdx.x & 31;
    int waveIb = threadIdx.x >> 5;
    int tile   = blockIdx.x * (blockDim.x >> 5) + waveIb;
    int ntiles = N >> 4;
    int mt = tile / ntiles;
    int nt = tile - mt * ntiles;
    int m0 = mt << 4, n0 = nt << 4;

    int mrow = m0 + (lane & 15);            // A row this lane carries
    int nrow = n0 + (lane & 15);            // B row (output column) this lane carries
    int kbA  = (lane >> 4) * 16;            // A K sub-base (lanes 16-31 -> +16)
    int kbB  = (lane >> 4) * 16;            // B K sub-base (lanes 16-31 -> +16)

    const bf16* arow = A + (size_t)mrow * K;
    const bf16* brow = B + (size_t)nrow * K;

    v8f c = {};
    for (int k0 = 0; k0 < K; k0 += 32) {
        v8bf alo = *(const v8bf*)(arow + k0 + kbA);        // K = kbA .. kbA+7
        v8bf ahi = *(const v8bf*)(arow + k0 + 8 + kbA);    // K = 8+kbA .. +7
        v16bf a;
        #pragma unroll
        for (int j = 0; j < 8; ++j) { a[j] = alo[j]; a[8 + j] = ahi[j]; }
        v16bf b = *(const v16bf*)(brow + k0 + kbB);        // K = kbB .. kbB+15
        c = __builtin_amdgcn_wmma_f32_16x16x32_bf16(false, a, false, b,
                                                    (short)0, c, false, false);
    }
    // C/D layout: lane<16 -> M = m0+v, lane>=16 -> M = m0+8+v ; N = n0+(lane&15)
    int ncol  = n0 + (lane & 15);
    int mbase = m0 + ((lane >> 4) << 3);
    #pragma unroll
    for (int v = 0; v < 8; ++v)
        D[(size_t)(mbase + v) * N + ncol] = c[v];
}

// ---------------- bias + row L2-normalize * 1.8 -> bf16 ----------------
__global__ void row_norm_kernel(const float* __restrict__ hraw, const float* __restrict__ b2,
                                bf16* __restrict__ hnb) {
    __shared__ float sm[CHC];
    int row = blockIdx.x, t = threadIdx.x;
    float v = hraw[(size_t)row * CHC + t] + b2[t];
    sm[t] = v * v;
    __syncthreads();
    for (int s = CHC / 2; s > 0; s >>= 1) {
        if (t < s) sm[t] += sm[t + s];
        __syncthreads();
    }
    float scale = 1.8f / fmaxf(sqrtf(sm[0]), 1e-12f);
    hnb[(size_t)row * CHC + t] = (bf16)(v * scale);
}

// ---------------- degree (self-loop = 1) ----------------
__global__ void deg_init_kernel(float* __restrict__ deg) {
    int i = blockIdx.x * 256 + threadIdx.x;
    if (i < N_NODESC) deg[i] = 1.0f;
}
__global__ void deg_scatter_kernel(const int* __restrict__ ei, float* __restrict__ deg) {
    int e = blockIdx.x * 256 + threadIdx.x;   // grid sized exactly to N_EDGES
    atomicAdd(deg + ei[N_EDGESC + e], 1.0f);
}
__global__ void dinv_kernel(const float* __restrict__ deg, float* __restrict__ dinv) {
    int i = blockIdx.x * 256 + threadIdx.x;
    if (i < N_NODESC) dinv[i] = rsqrtf(deg[i]);   // deg >= 1 always
}

// ---------------- y = xw * dinv[node]; acc = 0 ----------------
__global__ void y_acc_kernel(const float* __restrict__ xw, const float* __restrict__ dinv,
                             float* __restrict__ y, float* __restrict__ acc) {
    int i = blockIdx.x * 256 + threadIdx.x;        // grid covers N_NODES*CH exactly
    float d = dinv[i >> 7];
    y[i] = xw[i] * d;
    acc[i] = 0.0f;
}

// ---------------- scatter-add y[r] into acc[c] over edges ----------------
__global__ void edge_scatter_kernel(const int* __restrict__ ei, const float* __restrict__ y,
                                    float* __restrict__ acc) {
    int e  = blockIdx.x * 2 + (threadIdx.x >> 7);
    int ch = threadIdx.x & 127;
    int r = ei[e], c = ei[N_EDGESC + e];
    atomicAdd(acc + (size_t)c * CHC + ch, y[(size_t)r * CHC + ch]);
}

// ---------------- z1 = dinv[c]*(acc + y) + bg ----------------
__global__ void z1_fin_kernel(const float* __restrict__ acc, const float* __restrict__ y,
                              const float* __restrict__ dinv, const float* __restrict__ bg,
                              float* __restrict__ z1) {
    int i = blockIdx.x * 256 + threadIdx.x;
    int node = i >> 7, ch = i & 127;
    z1[i] = dinv[node] * (acc[i] + y[i]) + bg[ch];
}

// ---------------- z2 col0: per-row 2-way dot over 10000 + l2norm * 0.8 ----------------
__global__ void z2_rows_kernel(const float* __restrict__ x2, const float* __restrict__ W22,
                               float* __restrict__ z2c0) {
    __shared__ float s0[256], s1[256];
    int row = blockIdx.x, t = threadIdx.x;
    const float4* xr = (const float4*)(x2 + (size_t)row * N_NODESC);
    const float4* w0 = (const float4*)(W22);
    const float4* w1 = (const float4*)(W22 + N_NODESC);
    float a0 = 0.f, a1 = 0.f;
    for (int j = t; j < N_NODESC / 4; j += 256) {
        float4 xv = xr[j], v0 = w0[j], v1 = w1[j];
        a0 += xv.x * v0.x + xv.y * v0.y + xv.z * v0.z + xv.w * v0.w;
        a1 += xv.x * v1.x + xv.y * v1.y + xv.z * v1.z + xv.w * v1.w;
    }
    s0[t] = a0; s1[t] = a1;
    __syncthreads();
    for (int s = 128; s > 0; s >>= 1) {
        if (t < s) { s0[t] += s0[t + s]; s1[t] += s1[t + s]; }
        __syncthreads();
    }
    if (t == 0) {
        float n = sqrtf(s0[0] * s0[0] + s1[0] * s1[0]);
        z2c0[row] = 0.8f * s0[0] / fmaxf(n, 1e-12f);
    }
}

// ---------------- per-edge output: wave32 per edge ----------------
__global__ void edge_out_kernel(const int* __restrict__ ei, const float* __restrict__ z1,
                                const float* __restrict__ z2c0, float* __restrict__ out) {
    int lane = threadIdx.x & 31;
    int wave = threadIdx.x >> 5;
    int e = blockIdx.x * 8 + wave;            // grid sized exactly to N_EDGES/8
    int r = ei[e], c = ei[N_EDGESC + e];
    float4 a = ((const float4*)(z1 + (size_t)r * CHC))[lane];
    float4 b = ((const float4*)(z1 + (size_t)c * CHC))[lane];
    float p = a.x * b.x + a.y * b.y + a.z * b.z + a.w * b.w;
    #pragma unroll
    for (int off = 16; off > 0; off >>= 1) p += __shfl_xor(p, off, 32);
    if (lane == 0) {
        float sf = 1.0f / (1.0f + __expf(-p));
        float vn = z2c0[r] + z2c0[c];
        float sn = 1.0f / (1.0f + __expf(-vn));
        out[e] = sf * sf + (1.0f - sf) * sn;
    }
}

extern "C" void kernel_launch(void* const* d_in, const int* in_sizes, int n_in,
                              void* d_out, int out_size, void* d_ws, size_t ws_size,
                              hipStream_t stream) {
    const float* x   = (const float*)d_in[0];   // [10000,512]
    const float* x2  = (const float*)d_in[1];   // [10000,10000]
    const float* W2  = (const float*)d_in[2];   // [128,512]
    const float* b2  = (const float*)d_in[3];   // [128]
    const float* Wg  = (const float*)d_in[4];   // [128,128]
    const float* bg  = (const float*)d_in[5];   // [128]
    const float* W22 = (const float*)d_in[6];   // [2,10000]
    const int*   ei  = (const int*)d_in[7];     // [2,320000] (int64 in ref but JAX x64-off -> int32)
    float* out = (float*)d_out;

    // carve workspace (256B aligned slices)
    char* wsp = (char*)d_ws;
    auto carve = [&](size_t bytes) -> void* {
        void* p = (void*)wsp;
        wsp += (bytes + 255) & ~(size_t)255;
        return p;
    };
    bf16*  xb   = (bf16*) carve((size_t)N_NODESC * IN_DIMC * sizeof(bf16));
    bf16*  w2b  = (bf16*) carve((size_t)CHC * IN_DIMC * sizeof(bf16));
    bf16*  wgb  = (bf16*) carve((size_t)CHC * CHC * sizeof(bf16));
    float* hraw = (float*)carve((size_t)N_NODESC * CHC * sizeof(float));
    bf16*  hnb  = (bf16*) carve((size_t)N_NODESC * CHC * sizeof(bf16));
    float* xw   = (float*)carve((size_t)N_NODESC * CHC * sizeof(float));
    float* y    = (float*)carve((size_t)N_NODESC * CHC * sizeof(float));
    float* acc  = (float*)carve((size_t)N_NODESC * CHC * sizeof(float));
    float* z1   = (float*)carve((size_t)N_NODESC * CHC * sizeof(float));
    float* deg  = (float*)carve((size_t)N_NODESC * sizeof(float));
    float* dinv = (float*)carve((size_t)N_NODESC * sizeof(float));
    float* z2c0 = (float*)carve((size_t)N_NODESC * sizeof(float));

    // 1) bf16 conversions
    cvt_bf16_kernel<<<(N_NODESC * IN_DIMC) / 256, 256, 0, stream>>>(x, xb, N_NODESC * IN_DIMC);
    cvt_bf16_kernel<<<(CHC * IN_DIMC) / 256, 256, 0, stream>>>(W2, w2b, CHC * IN_DIMC);
    cvt_bf16_kernel<<<(CHC * CHC) / 256, 256, 0, stream>>>(Wg, wgb, CHC * CHC);

    // 2) h = x @ W2^T   (WMMA)  : tiles = 625*8 = 5000; 4 waves/block
    wmma_gemm_kernel<<<5000 / 4, 128, 0, stream>>>(xb, w2b, hraw, N_NODESC, CHC, IN_DIMC);

    // 3) h = l2norm(h + b2) * 1.8  -> bf16
    row_norm_kernel<<<N_NODESC, CHC, 0, stream>>>(hraw, b2, hnb);

    // 4) xw = h_norm @ Wg^T (WMMA)
    wmma_gemm_kernel<<<5000 / 4, 128, 0, stream>>>(hnb, wgb, xw, N_NODESC, CHC, CHC);

    // 5) degrees + dinv
    deg_init_kernel<<<(N_NODESC + 255) / 256, 256, 0, stream>>>(deg);
    deg_scatter_kernel<<<N_EDGESC / 256, 256, 0, stream>>>(ei, deg);
    dinv_kernel<<<(N_NODESC + 255) / 256, 256, 0, stream>>>(deg, dinv);

    // 6) y = xw*dinv, acc = 0 ; scatter ; z1 = dinv*(acc+y)+bg
    y_acc_kernel<<<(N_NODESC * CHC) / 256, 256, 0, stream>>>(xw, dinv, y, acc);
    edge_scatter_kernel<<<N_EDGESC / 2, 256, 0, stream>>>(ei, y, acc);
    z1_fin_kernel<<<(N_NODESC * CHC) / 256, 256, 0, stream>>>(acc, y, dinv, bg, z1);

    // 7) z2 column 0 (bandwidth-bound 400MB stream of x2)
    z2_rows_kernel<<<N_NODESC, 256, 0, stream>>>(x2, W22, z2c0);

    // 8) per-edge decode
    edge_out_kernel<<<N_EDGESC / 8, 256, 0, stream>>>(ei, z1, z2c0, out);
}
